// DotProductAttention_62801011802725
// MI455X (gfx1250) — matrix operand order repaired and checked by reading, not compiled, not run on hardware
//
#include <hip/hip_runtime.h>
#include <hip/hip_bf16.h>
#include <stdint.h>

typedef __attribute__((ext_vector_type(16))) __bf16 v16bf;
typedef __attribute__((ext_vector_type(2)))  __bf16 v2bf;
typedef __attribute__((ext_vector_type(8)))  float  v8f;

#define S_LEN   2048
#define D_DIM   512
#define KB      32      // keys per iteration
#define M_TILE  64      // query rows per workgroup

// Padded LDS row strides (64 banks x 4B): shift rows off bank-aligned columns.
#define QKROWB  (D_DIM * 2 + 16)   // 1040 B -> bank shift 4/row: 16 clean groups
#define VTROWB  (KB * 2 + 16)      // 80 B   -> bank shift 20/row
#define PROWB   (KB * 2 + 16)      // 80 B

// LDS layout (bytes)
#define Q_OFF    0
#define Q_BYTES  (M_TILE * QKROWB)        // 66560
#define K_OFF    (Q_OFF + Q_BYTES)
#define K_BYTES  (KB * QKROWB)            // 33280
#define VT_OFF   (K_OFF + K_BYTES)
#define VT_BYTES (D_DIM * VTROWB)         // 40960 (V transposed: [d][key])
#define P_OFF    (VT_OFF + VT_BYTES)
#define P_BYTES  (4 * 16 * PROWB)         // 5120  (per wave-pair P tile)
#define SUM_OFF  (P_OFF + P_BYTES)
#define SUM_BYTES (4 * 2 * 16 * 4)        // 512
#define SMEM_BYTES (SUM_OFF + SUM_BYTES)  // 146432

#if defined(__gfx1250__) && __has_builtin(__builtin_amdgcn_cvt_pk_bf16_f32)
__device__ __forceinline__ unsigned pk2bf(float a, float b) {
  union { v2bf v; unsigned u; } c;
  c.v = __builtin_amdgcn_cvt_pk_bf16_f32(a, b);   // 1 VALU op, RNE
  return c.u;
}
#else
// Native casts: clang legalizes fptrunc f32->bf16 to v_cvt_pk_bf16_f32 /
// v_cvt_bf16_f32 when available; correct RNE software sequence otherwise.
__device__ __forceinline__ unsigned pk2bf(float a, float b) {
  union { __bf16 h[2]; unsigned u; } c;
  c.h[0] = (__bf16)a;
  c.h[1] = (__bf16)b;
  return c.u;
}
#endif

union Frag16 { v16bf v; uint4 q[2]; };

__device__ __forceinline__ v16bf ld_frag(const char* p0, const char* p1) {
  Frag16 f;
  f.q[0] = *(const uint4*)p0;   // ds_load_b128
  f.q[1] = *(const uint4*)p1;   // ds_load_b128
  return f.v;
}

__global__ __launch_bounds__(256, 4)    // cap <=256 VGPRs: no vgpr-msb juggling
void attn_fa_kernel(const float* __restrict__ qg,
                    const float* __restrict__ kg,
                    const float* __restrict__ vg,
                    const int*   __restrict__ valid_lens,
                    float*       __restrict__ outg) {
  extern __shared__ char smem[];
  char*  Qb = smem + Q_OFF;
  char*  Kb = smem + K_OFF;
  char*  Vt = smem + VT_OFF;
  char*  Pb = smem + P_OFF;
  float* Sm = (float*)(smem + SUM_OFF);

  const int n    = blockIdx.y;
  const int q0   = blockIdx.x * M_TILE;
  const int t    = threadIdx.x;
  const int wave = t >> 5, lane = t & 31;
  const int pair = wave >> 1, h = wave & 1;   // pair shares 16 q-rows; h = key/output half
  const int lq   = lane & 15, lh = lane >> 4;

  const int  Lraw    = valid_lens[n];
  const bool uniform = (Lraw == 0);           // all-masked -> uniform softmax
  const int  L       = uniform ? S_LEN : Lraw;
  const int  nblocks = (L + KB - 1) / KB;
  const float scale_eff = uniform ? 0.0f : 0.044194173824159216f; // 1/sqrt(512)

  // ---- stage Q tile (M_TILE x D) row-major bf16 into LDS (once) ----
  {
    const float* qp = qg + ((size_t)n * S_LEN + q0) * D_DIM;
    #pragma unroll 4
    for (int j = 0; j < 16; ++j) {            // 4096 float8 chunks / 256 thr
      int i   = j * 256 + t;
      int row = i >> 6;                       // D/8 = 64 chunks per row
      int c8  = i & 63;
      const float* p = qp + row * D_DIM + c8 * 8;
      float4 a = *(const float4*)p;
      float4 b = *(const float4*)(p + 4);
      *(uint4*)(Qb + row * QKROWB + c8 * 16) =
          make_uint4(pk2bf(a.x, a.y), pk2bf(a.z, a.w),
                     pk2bf(b.x, b.y), pk2bf(b.z, b.w));
    }
  }

  v8f acc[16];                                // 16 q-rows x 256 d-cols of O
  #pragma unroll
  for (int i = 0; i < 16; ++i) acc[i] = (v8f)(0.0f);
  float rowsum = 0.0f;

  for (int kb = 0; kb < nblocks; ++kb) {
    __syncthreads();                          // prior iter's LDS consumers done

    const float* kp = kg + ((size_t)n * S_LEN + (size_t)kb * KB) * D_DIM;
    const float* vp = vg + ((size_t)n * S_LEN + (size_t)kb * KB) * D_DIM;

    // ---- K block: row-major bf16, coalesced loads, conflict-free b128 stores
    #pragma unroll 4
    for (int j = 0; j < 8; ++j) {             // 2048 float8 chunks / 256 thr
      int i   = j * 256 + t;
      int row = i >> 6;
      int c8  = i & 63;
      const float* p = kp + row * D_DIM + c8 * 8;
      float4 a = *(const float4*)p;
      float4 b = *(const float4*)(p + 4);
      *(uint4*)(Kb + row * QKROWB + c8 * 16) =
          make_uint4(pk2bf(a.x, a.y), pk2bf(a.z, a.w),
                     pk2bf(b.x, b.y), pk2bf(b.z, b.w));
    }

    // ---- V block transposed: thread t owns dims d = t, t+256 (all 32 keys).
    //      Single base pointer per half; key strides are compile-time byte
    //      offsets (kk*4096 < 24-bit imm) -> no per-load address math.
    //      Per key row, lanes read 128B contiguous (coalesced b32);
    //      whole VT row written as 4 conflict-free b128 stores.
    #pragma unroll
    for (int half = 0; half < 2; ++half) {
      const int d = t + half * 256;
      const float* vpd = vp + d;
      unsigned pk[16];
      #pragma unroll
      for (int kk = 0; kk < 16; ++kk)
        pk[kk] = pk2bf(vpd[(2 * kk)     * D_DIM],
                       vpd[(2 * kk + 1) * D_DIM]);
      char* dst = Vt + d * VTROWB;
      *(uint4*)(dst)      = make_uint4(pk[0],  pk[1],  pk[2],  pk[3]);
      *(uint4*)(dst + 16) = make_uint4(pk[4],  pk[5],  pk[6],  pk[7]);
      *(uint4*)(dst + 32) = make_uint4(pk[8],  pk[9],  pk[10], pk[11]);
      *(uint4*)(dst + 48) = make_uint4(pk[12], pk[13], pk[14], pk[15]);
    }

    if (kb + 1 < nblocks) {                   // warm L2 for next block
      __builtin_prefetch(kp + KB * D_DIM + (size_t)t * 64, 0, 1);
      __builtin_prefetch(vp + KB * D_DIM + (size_t)t * 64, 0, 1);
    }
    __syncthreads();

    // ---- S^T(half h) = K_half(16xD) x Q^T(Dx16), fp32 accum over D ----
    v8f cs = (v8f)(0.0f);
    const char* krow = Kb + (h * 16 + lq) * QKROWB;        // A: M=key
    const char* qrow = Qb + (pair * 16 + lq) * QKROWB;     // B: N=query
    #pragma unroll
    for (int s = 0; s < 16; ++s) {
      v16bf a = ld_frag(krow + s * 64 + lh * 16,           // A chunks {lh*8,16+lh*8}
                        krow + s * 64 + lh * 16 + 32);
      v16bf b = ld_frag(qrow + s * 64 + lh * 16,           // B chunks {lh*8,16+lh*8}
                        qrow + s * 64 + lh * 16 + 32);
      cs = __builtin_amdgcn_wmma_f32_16x16x32_bf16(false, a, false, b,
                                                   (short)0, cs, false, false);
    }

    // ---- numerator: p = exp(score/sqrt(D)) with tail mask; track row sums ----
    float p[8];
    #pragma unroll
    for (int r = 0; r < 8; ++r) {
      int key = kb * KB + h * 16 + lh * 8 + r;             // S^T: M=key, N=query
      p[r] = (key < L) ? __expf(cs[r] * scale_eff) : 0.0f;
      rowsum += p[r];
    }

    // ---- P (bf16) into the pair's LDS tile: row=query, col=key-in-block ----
    *(uint4*)(Pb + pair * (16 * PROWB) + lq * PROWB + h * 32 + lh * 16) =
        make_uint4(pk2bf(p[0], p[1]), pk2bf(p[2], p[3]),
                   pk2bf(p[4], p[5]), pk2bf(p[6], p[7]));
    __syncthreads();                                       // pair halves visible

    // ---- O(16 x 256-half) += P(16xKB) x V(KB x D) ----
    const char* prow = Pb + pair * (16 * PROWB) + lq * PROWB;
    v16bf pa = ld_frag(prow + lh * 16, prow + 32 + lh * 16);
    const char* vtb = Vt + (h * 256 + lq) * VTROWB + lh * 16;
    #pragma unroll
    for (int dt = 0; dt < 16; ++dt) {
      v16bf vb = ld_frag(vtb + dt * 16 * VTROWB,
                         vtb + dt * 16 * VTROWB + 32);
      acc[dt] = __builtin_amdgcn_wmma_f32_16x16x32_bf16(false, pa, false, vb,
                                                        (short)0, acc[dt], false, false);
    }
  }

  // ---- finalize: combine row sums across lane halves + wave pair ----
  rowsum += __shfl_xor(rowsum, 16);
  if (lane < 16) Sm[pair * 32 + h * 16 + lane] = rowsum;
  __syncthreads();

  float inv[8];
  #pragma unroll
  for (int r = 0; r < 8; ++r) {
    int m = lh * 8 + r;                        // C layout: M = r + 8*lh
    inv[r] = 1.0f / (Sm[pair * 32 + m] + Sm[pair * 32 + 16 + m]);
  }

  float* op = outg + ((size_t)n * S_LEN + q0 + pair * 16) * D_DIM + h * 256;
  #pragma unroll
  for (int dt = 0; dt < 16; ++dt) {
    #pragma unroll
    for (int r = 0; r < 8; ++r) {
      int m = lh * 8 + r;
      op[(size_t)m * D_DIM + dt * 16 + lq] = acc[dt][r] * inv[r];
    }
  }
}

extern "C" void kernel_launch(void* const* d_in, const int* in_sizes, int n_in,
                              void* d_out, int out_size, void* d_ws, size_t ws_size,
                              hipStream_t stream) {
  const float* q  = (const float*)d_in[0];
  const float* k  = (const float*)d_in[1];
  const float* v  = (const float*)d_in[2];
  const int*   vl = (const int*)d_in[3];
  float* out = (float*)d_out;

  const int N = in_sizes[3];
  static_assert(SMEM_BYTES <= 320 * 1024, "fits WGP LDS");
  hipFuncSetAttribute((const void*)attn_fa_kernel,
                      hipFuncAttributeMaxDynamicSharedMemorySize, SMEM_BYTES);

  dim3 grid(S_LEN / M_TILE, N);
  dim3 block(256);
  hipLaunchKernelGGL(attn_fa_kernel, grid, block, SMEM_BYTES, stream,
                     q, k, v, vl, out);
}
